// NeuronGAT_54296976556689
// MI455X (gfx1250) — matrix operand order repaired and checked by reading, not compiled, not run on hardware
//
#include <hip/hip_runtime.h>
#include <math.h>

// ---------------------------------------------------------------------------
// NeuronGAT forward for MI455X (gfx1250, wave32, WMMA).
// GEMMs: v_wmma_f32_16x16x32_f16, 64x64 block tile, 4 waves, each wave owns a
// 16x64 output strip (A-fragment reused across 4 B-fragments -> 4 WMMA/kstep).
// Self-attention: flash-attention; K/V tiles staged into LDS with
// GLOBAL_LOAD_ASYNC_TO_LDS_B128 (ASYNCcnt) and converted f32->f16 at
// fragment-build time; 8 WMMA per 32-key block per wave.
// All 32-bit index arithmetic (products < 2^31) to avoid v_mul_u64 chains.
// ---------------------------------------------------------------------------

typedef __attribute__((ext_vector_type(16))) _Float16 v16h;
typedef __attribute__((ext_vector_type(4)))  _Float16 v4h;
typedef __attribute__((ext_vector_type(8)))  float    v8f;

#define NNODE 4096
#define EEDGE 131072
#define ETOT  (EEDGE + NNODE)
#define HID   256
#define GHEADS 8
#define GHD    32
#define SAHEADS 4
#define SAHD    64
#define NEG_SLOPE 0.2f

#define WMMA_F16(A_, B_, C_) \
  __builtin_amdgcn_wmma_f32_16x16x32_f16(false, (A_), false, (B_), (short)0, (C_), false, false)

__device__ __forceinline__ float gelu_f(float x) {
  return 0.5f * x * (1.0f + erff(x * 0.70710678118654752f));
}
__device__ __forceinline__ float lrelu_f(float x) { return x > 0.f ? x : NEG_SLOPE * x; }

__device__ __forceinline__ unsigned fenc(float f) {
  unsigned u = __float_as_uint(f);
  return (u & 0x80000000u) ? ~u : (u | 0x80000000u);
}
__device__ __forceinline__ float fdec(unsigned e) {
  return (e & 0x80000000u) ? __uint_as_float(e & 0x7fffffffu) : __uint_as_float(~e);
}

__device__ __forceinline__ v4h cvt4(float4 f) {
  v4h h = { (_Float16)f.x, (_Float16)f.y, (_Float16)f.z, (_Float16)f.w };
  return h;
}

// Per ISA 10.2: generic (flat) LDS address maps to LDS via addr[31:0].
__device__ __forceinline__ unsigned lds_addr32(const void* p) {
  return (unsigned)(unsigned long long)p;
}
// Async DMA: 16B per lane, global -> LDS, tracked with ASYNCcnt.
__device__ __forceinline__ void async_ld_b128(unsigned ldsOff, const float* g) {
  asm volatile("global_load_async_to_lds_b128 %0, %1, off" :: "v"(ldsOff), "v"(g) : "memory");
}
__device__ __forceinline__ void wait_async0() {
  asm volatile("s_wait_asynccnt 0" ::: "memory");
}

// ---------------------------------------------------------------------------
__global__ void k_fill(float* p, float v, int n) {
  int i = blockIdx.x * blockDim.x + threadIdx.x;
  if (i < n) p[i] = v;
}

__global__ void k_edges(const int* __restrict__ ei, int* __restrict__ src, int* __restrict__ dst) {
  int i = blockIdx.x * blockDim.x + threadIdx.x;
  if (i >= ETOT) return;
  if (i < EEDGE) { src[i] = ei[i]; dst[i] = ei[EEDGE + i]; }
  else           { src[i] = i - EEDGE; dst[i] = i - EEDGE; }
}

// ---------------------------------------------------------------------------
// C[M,Nn] = A[M,K] * W[Nn,K]^T + bias.  Grid (Nn/64, M/64), 128 thr = 4 waves.
// ---------------------------------------------------------------------------
__global__ __launch_bounds__(128)
void k_gemm(const float* __restrict__ A, const float* __restrict__ W,
            const float* __restrict__ bias, float* __restrict__ C,
            int M, int Nn, int K) {
  __shared__ _Float16 sA[64][32];
  __shared__ _Float16 sB[64][32];
  const int tid = threadIdx.x;
  const int wave = tid >> 5, lane = tid & 31;
  const int half = lane >> 4, l16 = lane & 15;
  const int tN = blockIdx.x * 64, tM = blockIdx.y * 64;
  const int rL = tid >> 1;            // LDS fill row
  const int cL = (tid & 1) * 16;      // col base (16 cols per thread)
  v8f acc0 = {}, acc1 = {}, acc2 = {}, acc3 = {};

  for (int k0 = 0; k0 < K; k0 += 32) {
    const float* pa = &A[(tM + rL) * K + k0 + cL];
    const float* pb = &W[(tN + rL) * K + k0 + cL];
#pragma unroll
    for (int u = 0; u < 4; ++u)
      *(v4h*)&sA[rL][cL + 4 * u] = cvt4(*(const float4*)&pa[4 * u]);
#pragma unroll
    for (int u = 0; u < 4; ++u)
      *(v4h*)&sB[rL][cL + 4 * u] = cvt4(*(const float4*)&pb[4 * u]);
    __builtin_prefetch(&pa[32], 0, 1);  // global_prefetch_b8 (next K tile)
    __syncthreads();

    v16h af;
#pragma unroll
    for (int v = 0; v < 8; ++v) {  // A fragment: contiguous k = 16*half + s, same order as B
      int ka = 16 * half + 2 * v;
      af[2 * v]     = sA[wave * 16 + l16][ka];
      af[2 * v + 1] = sA[wave * 16 + l16][ka + 1];
    }
#define GEMM_NT(ACC, NT)                                                    \
    { v16h bf;                                                              \
      _Pragma("unroll")                                                     \
      for (int v = 0; v < 8; ++v) { int kb = 16 * half + 2 * v;             \
        bf[2*v]   = sB[(NT) * 16 + l16][kb];                                \
        bf[2*v+1] = sB[(NT) * 16 + l16][kb + 1]; }                          \
      ACC = WMMA_F16(af, bf, ACC); }
    GEMM_NT(acc0, 0)
    GEMM_NT(acc1, 1)
    GEMM_NT(acc2, 2)
    GEMM_NT(acc3, 3)
#undef GEMM_NT
    __syncthreads();
  }

#define GEMM_ST(ACC, NT)                                                    \
  { _Pragma("unroll")                                                       \
    for (int v = 0; v < 8; ++v) {                                           \
      int r = tM + wave * 16 + v + 8 * half;                                \
      int c = tN + (NT) * 16 + l16;                                         \
      C[r * Nn + c] = ACC[v] + bias[c]; } }
  GEMM_ST(acc0, 0)
  GEMM_ST(acc1, 1)
  GEMM_ST(acc2, 2)
  GEMM_ST(acc3, 3)
#undef GEMM_ST
}

// ---------------------------------------------------------------------------
// LayerNorm over 256 channels + activation (0=none,1=gelu,2=leaky).
// ---------------------------------------------------------------------------
__global__ __launch_bounds__(256)
void k_ln_act(const float* __restrict__ X, const float* __restrict__ g,
              const float* __restrict__ b, float* __restrict__ out,
              float* __restrict__ out2, int out2Stride, int out2Ofs, int act) {
  int wave = threadIdx.x >> 5, lane = threadIdx.x & 31;
  int row = blockIdx.x * 8 + wave;
  float x[8], s = 0.f, ss = 0.f;
#pragma unroll
  for (int j = 0; j < 8; ++j) {
    x[j] = X[row * HID + lane + 32 * j];
    s += x[j]; ss += x[j] * x[j];
  }
#pragma unroll
  for (int off = 16; off >= 1; off >>= 1) { s += __shfl_xor(s, off); ss += __shfl_xor(ss, off); }
  float mu = s * (1.f / 256.f);
  float rs = rsqrtf(ss * (1.f / 256.f) - mu * mu + 1e-5f);
#pragma unroll
  for (int j = 0; j < 8; ++j) {
    int c = lane + 32 * j;
    float y = (x[j] - mu) * rs * g[c] + b[c];
    if (act == 1) y = gelu_f(y); else if (act == 2) y = lrelu_f(y);
    out[row * HID + c] = y;
    if (out2) out2[row * out2Stride + out2Ofs + c] = y;
  }
}

__global__ void k_act(const float* __restrict__ x, float* __restrict__ y, int n, int act) {
  int i = blockIdx.x * blockDim.x + threadIdx.x;
  if (i >= n) return;
  float v = x[i];
  y[i] = (act == 1) ? gelu_f(v) : (act == 2 ? lrelu_f(v) : v);
}

__global__ void k_add2(const float* __restrict__ a, const float* __restrict__ b,
                       float* __restrict__ out, int outStride, int outOfs, int n) {
  int i = blockIdx.x * blockDim.x + threadIdx.x;
  if (i >= n) return;
  int r = i >> 8, c = i & 255;
  out[r * outStride + outOfs + c] = a[i] + b[i];
}

// ---------------------------------------------------------------------------
// GATv2 edge passes
// ---------------------------------------------------------------------------
__global__ void k_gat_logits(const float* __restrict__ xl, const float* __restrict__ xr,
                             const float* __restrict__ att, const int* __restrict__ src,
                             const int* __restrict__ dst, float* __restrict__ logits,
                             unsigned* __restrict__ menc) {
  int idx = blockIdx.x * blockDim.x + threadIdx.x;
  if (idx >= ETOT * GHEADS) return;
  int e = idx >> 3, h = idx & 7;
  int s = src[e], d = dst[e];
  const float* pl = xl + s * HID + h * GHD;
  const float* pr = xr + d * HID + h * GHD;
  const float* pa = att + h * GHD;
  float acc = 0.f;
#pragma unroll
  for (int u = 0; u < 8; ++u) {
    float4 a4 = *(const float4*)&pl[4 * u];
    float4 b4 = *(const float4*)&pr[4 * u];
    float4 w4 = *(const float4*)&pa[4 * u];
    acc += lrelu_f(a4.x + b4.x) * w4.x + lrelu_f(a4.y + b4.y) * w4.y +
           lrelu_f(a4.z + b4.z) * w4.z + lrelu_f(a4.w + b4.w) * w4.w;
  }
  logits[idx] = acc;
  atomicMax(&menc[d * GHEADS + h], fenc(acc));
}

__global__ void k_gat_expsum(const float* __restrict__ logits, const unsigned* __restrict__ menc,
                             const int* __restrict__ dst, float* __restrict__ a,
                             float* __restrict__ denom) {
  int idx = blockIdx.x * blockDim.x + threadIdx.x;
  if (idx >= ETOT * GHEADS) return;
  int e = idx >> 3, h = idx & 7;
  int d = dst[e];
  float v = __expf(logits[idx] - fdec(menc[d * GHEADS + h]));
  a[idx] = v;
  atomicAdd(&denom[d * GHEADS + h], v);
}

__global__ __launch_bounds__(256)
void k_gat_scatter(const float* __restrict__ a, const float* __restrict__ xl,
                   const int* __restrict__ src, const int* __restrict__ dst,
                   float* __restrict__ gacc) {
  int e = blockIdx.x, c = threadIdx.x;
  int s = src[e], d = dst[e], h = c >> 5;
  atomicAdd(&gacc[d * HID + c], a[e * GHEADS + h] * xl[s * HID + c]);
}

__global__ void k_gat_final(const float* __restrict__ gacc, const float* __restrict__ denom,
                            const float* __restrict__ bias, float* __restrict__ out) {
  int i = blockIdx.x * blockDim.x + threadIdx.x;
  if (i >= NNODE * HID) return;
  int n = i >> 8, c = i & 255;
  out[i] = gacc[i] / denom[n * GHEADS + (c >> 5)] + bias[c];
}

// ---------------------------------------------------------------------------
// SE layer
// ---------------------------------------------------------------------------
__global__ __launch_bounds__(256)
void k_colmean(const float* __restrict__ x, float* __restrict__ mean) {
  __shared__ float red[256];
  int c = blockIdx.x, tid = threadIdx.x;
  float s = 0.f;
  for (int r = tid; r < NNODE; r += 256) s += x[r * HID + c];
  red[tid] = s; __syncthreads();
  for (int w = 128; w > 0; w >>= 1) { if (tid < w) red[tid] += red[tid + w]; __syncthreads(); }
  if (tid == 0) mean[c] = red[0] * (1.f / (float)NNODE);
}

__global__ __launch_bounds__(256)
void k_se_mlp(const float* __restrict__ mean, const float* __restrict__ w1,
              const float* __restrict__ w2, float* __restrict__ gate) {
  __shared__ float hid[32];
  int tid = threadIdx.x;
  if (tid < 32) {
    float s = 0.f;
    for (int c = 0; c < 256; ++c) s += mean[c] * w1[tid * 256 + c];
    hid[tid] = fmaxf(s, 0.f);
  }
  __syncthreads();
  float s = 0.f;
#pragma unroll
  for (int j = 0; j < 32; ++j) s += hid[j] * w2[tid * 32 + j];
  gate[tid] = 1.f / (1.f + __expf(-s));
}

__global__ void k_se_res(const float* __restrict__ x, const float* __restrict__ gate,
                         const float* __restrict__ res, float* __restrict__ out) {
  int i = blockIdx.x * blockDim.x + threadIdx.x;
  if (i >= NNODE * HID) return;
  out[i] = x[i] * gate[i & 255] + res[i];
}

// ---------------------------------------------------------------------------
// Flash attention: grid (N/64, 4 heads), 128 thr = 4 waves; wave = 16-row
// query tile, 16x64 f32 accumulator; key loop in blocks of 32.
// K/V tiles DMAed into LDS as raw f32 via GLOBAL_LOAD_ASYNC_TO_LDS_B128,
// completion enforced with s_wait_asynccnt 0 + workgroup barrier; f32->f16
// conversion folded into fragment builds.  8 WMMA per key block per wave.
// ---------------------------------------------------------------------------
__global__ __launch_bounds__(128)
void k_flash(const float* __restrict__ Qg, const float* __restrict__ Kg,
             const float* __restrict__ Vg, float* __restrict__ ctx) {
  __shared__ _Float16 sQ[64][64];
  __shared__ float    sKf[32][64];
  __shared__ float    sVf[32][64];
  __shared__ _Float16 sP[4][16][32];
  const int tid = threadIdx.x, wave = tid >> 5, lane = tid & 31;
  const int half = lane >> 4, l16 = lane & 15;
  const int q0 = blockIdx.x * 64;
  const int hofs = blockIdx.y * SAHD;
  const float scale = 0.125f;  // 1/sqrt(64) folded into Q

  {  // sQ fill: thread covers row tid>>1, 32 cols
    const int r = tid >> 1, cb = (tid & 1) * 32;
    const float* pq = &Qg[(q0 + r) * HID + hofs + cb];
#pragma unroll
    for (int u = 0; u < 8; ++u) {
      float4 f = *(const float4*)&pq[4 * u];
      v4h h = { (_Float16)(f.x * scale), (_Float16)(f.y * scale),
                (_Float16)(f.z * scale), (_Float16)(f.w * scale) };
      *(v4h*)&sQ[r][cb + 4 * u] = h;
    }
  }
  __syncthreads();

  v16h qa0, qa1;
#pragma unroll
  for (int v = 0; v < 8; ++v) {
    int ka = 16 * half + 2 * v;
    qa0[2 * v]     = sQ[wave * 16 + l16][ka];
    qa0[2 * v + 1] = sQ[wave * 16 + l16][ka + 1];
    qa1[2 * v]     = sQ[wave * 16 + l16][ka + 32];
    qa1[2 * v + 1] = sQ[wave * 16 + l16][ka + 33];
  }

  v8f O0 = {}, O1 = {}, O2 = {}, O3 = {};
  float m8[8], l8[8];
#pragma unroll
  for (int v = 0; v < 8; ++v) { m8[v] = -3.0e38f; l8[v] = 0.f; }

  const int rKV = tid >> 2, cKV = (tid & 3) * 16;  // 32 rows x 64 cols, 16B x4 per thread
  for (int kb = 0; kb < NNODE; kb += 32) {
    __syncthreads();  // previous tile fully consumed
    {
      const float* pk = &Kg[(kb + rKV) * HID + hofs + cKV];
      const float* pv = &Vg[(kb + rKV) * HID + hofs + cKV];
#pragma unroll
      for (int u = 0; u < 4; ++u)
        async_ld_b128(lds_addr32(&sKf[rKV][cKV + 4 * u]), pk + 4 * u);
#pragma unroll
      for (int u = 0; u < 4; ++u)
        async_ld_b128(lds_addr32(&sVf[rKV][cKV + 4 * u]), pv + 4 * u);
      wait_async0();
    }
    __syncthreads();

    v8f s0 = {}, s1 = {};
    {
      v16h bf;
#pragma unroll
      for (int v = 0; v < 8; ++v) { int kd = 16 * half + 2 * v;
        bf[2*v] = (_Float16)sKf[l16][kd];      bf[2*v+1] = (_Float16)sKf[l16][kd+1]; }
      s0 = WMMA_F16(qa0, bf, s0);
#pragma unroll
      for (int v = 0; v < 8; ++v) { int kd = 32 + 16 * half + 2 * v;
        bf[2*v] = (_Float16)sKf[l16][kd];      bf[2*v+1] = (_Float16)sKf[l16][kd+1]; }
      s0 = WMMA_F16(qa1, bf, s0);
#pragma unroll
      for (int v = 0; v < 8; ++v) { int kd = 16 * half + 2 * v;
        bf[2*v] = (_Float16)sKf[16+l16][kd];   bf[2*v+1] = (_Float16)sKf[16+l16][kd+1]; }
      s1 = WMMA_F16(qa0, bf, s1);
#pragma unroll
      for (int v = 0; v < 8; ++v) { int kd = 32 + 16 * half + 2 * v;
        bf[2*v] = (_Float16)sKf[16+l16][kd];   bf[2*v+1] = (_Float16)sKf[16+l16][kd+1]; }
      s1 = WMMA_F16(qa1, bf, s1);
    }

#pragma unroll
    for (int v = 0; v < 8; ++v) {
      float t = fmaxf(s0[v], s1[v]);
      t = fmaxf(t, __shfl_xor(t, 8)); t = fmaxf(t, __shfl_xor(t, 4));
      t = fmaxf(t, __shfl_xor(t, 2)); t = fmaxf(t, __shfl_xor(t, 1));
      float mn = fmaxf(m8[v], t);
      float corr = __expf(m8[v] - mn);
      float p0 = __expf(s0[v] - mn), p1 = __expf(s1[v] - mn);
      float rs = p0 + p1;
      rs += __shfl_xor(rs, 8); rs += __shfl_xor(rs, 4);
      rs += __shfl_xor(rs, 2); rs += __shfl_xor(rs, 1);
      l8[v] = l8[v] * corr + rs;
      m8[v] = mn;
      O0[v] *= corr; O1[v] *= corr; O2[v] *= corr; O3[v] *= corr;
      sP[wave][v + 8 * half][l16]      = (_Float16)p0;
      sP[wave][v + 8 * half][16 + l16] = (_Float16)p1;
    }

    v16h pa;
#pragma unroll
    for (int v = 0; v < 8; ++v) {
      int ka = 16 * half + 2 * v;
      pa[2 * v]     = sP[wave][l16][ka];
      pa[2 * v + 1] = sP[wave][l16][ka + 1];
    }
#define PV_STEP(OC, CBASE)                                                      \
    { v16h bv;                                                                  \
      _Pragma("unroll")                                                         \
      for (int v = 0; v < 8; ++v) { int kk = 16 * half + 2 * v;                 \
        bv[2*v]   = (_Float16)sVf[kk][(CBASE) + l16];                           \
        bv[2*v+1] = (_Float16)sVf[kk+1][(CBASE) + l16]; }                       \
      OC = WMMA_F16(pa, bv, OC); }
    PV_STEP(O0, 0)
    PV_STEP(O1, 16)
    PV_STEP(O2, 32)
    PV_STEP(O3, 48)
#undef PV_STEP
  }

#pragma unroll
  for (int v = 0; v < 8; ++v) {
    int r = q0 + wave * 16 + v + 8 * half;
    float inv = 1.f / l8[v];
    ctx[r * HID + hofs +  0 + l16] = O0[v] * inv;
    ctx[r * HID + hofs + 16 + l16] = O1[v] * inv;
    ctx[r * HID + hofs + 32 + l16] = O2[v] * inv;
    ctx[r * HID + hofs + 48 + l16] = O3[v] * inv;
  }
}

// final 128 -> 12 projection: grid (12, N/256), block 256
__global__ __launch_bounds__(256)
void k_head(const float* __restrict__ z, const float* __restrict__ w3,
            const float* __restrict__ b3, float* __restrict__ out) {
  int n = blockIdx.y * 256 + threadIdx.x;
  int o = blockIdx.x;
  const float* zr = z + n * 128;
  const float* wr = w3 + o * 128;
  float acc = b3[o];
#pragma unroll
  for (int u = 0; u < 32; ++u) {
    float4 a4 = *(const float4*)&zr[4 * u];
    float4 b4 = *(const float4*)&wr[4 * u];
    acc += a4.x * b4.x + a4.y * b4.y + a4.z * b4.z + a4.w * b4.w;
  }
  out[n * 12 + o] = acc;
}

// ---------------------------------------------------------------------------
extern "C" void kernel_launch(void* const* d_in, const int* in_sizes, int n_in,
                              void* d_out, int out_size, void* d_ws, size_t ws_size,
                              hipStream_t stream) {
  (void)in_sizes; (void)n_in; (void)out_size; (void)ws_size;
  const float* x  = (const float*)d_in[0];
  const int*   ei = (const int*)d_in[1];
  int p = 2;
  auto nxt = [&]() { return (const float*)d_in[p++]; };
  const float *ip_w = nxt(), *ip_b = nxt(), *ip_g = nxt(), *ip_bb = nxt();
  struct Blk { const float *wl,*bl,*wr,*br,*att,*bias,*se1,*se2,*g,*b,*resw; } blk[4];
  for (int i = 0; i < 4; ++i) {
    blk[i].wl = nxt(); blk[i].bl = nxt(); blk[i].wr = nxt(); blk[i].br = nxt();
    blk[i].att = nxt(); blk[i].bias = nxt(); blk[i].se1 = nxt(); blk[i].se2 = nxt();
    blk[i].g = nxt(); blk[i].b = nxt(); blk[i].resw = nxt();
  }
  const float *ln1g = nxt(), *ln1b = nxt(), *wq = nxt(), *bq = nxt(), *wk = nxt(), *bk = nxt();
  const float *wv = nxt(), *bvv = nxt(), *wo = nxt(), *bo = nxt(), *ln2g = nxt(), *ln2b = nxt();
  const float *fw1 = nxt(), *fb1 = nxt(), *fw2 = nxt(), *fb2 = nxt();
  const float *mw1 = nxt(), *mb1 = nxt(), *mg = nxt(), *mb = nxt();
  const float *mw2 = nxt(), *mb2 = nxt(), *mw3 = nxt(), *mb3 = nxt();

  char* ws = (char*)d_ws; size_t off = 0;
  auto carve = [&](size_t bytes) -> void* {
    void* q = ws + off; off = (off + bytes + 255) & ~(size_t)255; return q;
  };
  int*   src   = (int*)carve((size_t)ETOT * 4);
  int*   dst   = (int*)carve((size_t)ETOT * 4);
  float* h     = (float*)carve((size_t)NNODE * HID * 4);
  float* res   = (float*)carve((size_t)NNODE * HID * 4);
  float* xl    = (float*)carve((size_t)NNODE * HID * 4);
  float* xr    = (float*)carve((size_t)NNODE * HID * 4);
  float* gacc  = (float*)carve((size_t)NNODE * HID * 4);
  float* t2    = (float*)carve((size_t)NNODE * HID * 4);
  float* t3    = (float*)carve((size_t)NNODE * HID * 4);
  float* logit = (float*)carve((size_t)ETOT * GHEADS * 4);
  float* ae    = (float*)carve((size_t)ETOT * GHEADS * 4);
  unsigned* menc = (unsigned*)carve((size_t)NNODE * GHEADS * 4);
  float* denom = (float*)carve((size_t)NNODE * GHEADS * 4);
  float* mean  = (float*)carve(1024);
  float* gate  = (float*)carve(1024);
  float* zerob = (float*)carve(1024);
  float* jk    = (float*)carve((size_t)NNODE * 1024 * 4);
  float* qb    = (float*)carve((size_t)NNODE * HID * 4);
  float* kb2   = (float*)carve((size_t)NNODE * HID * 4);
  float* vb    = (float*)carve((size_t)NNODE * HID * 4);
  float* ctx   = (float*)carve((size_t)NNODE * HID * 4);
  float* t1    = (float*)carve((size_t)NNODE * 512 * 4);
  float* z2    = (float*)carve((size_t)NNODE * 128 * 4);

  const dim3 B256(256);
  auto GEMM = [&](const float* A, const float* W, const float* bias, float* C,
                  int M, int Nc, int K) {
    k_gemm<<<dim3(Nc / 64, M / 64), dim3(128), 0, stream>>>(A, W, bias, C, M, Nc, K);
  };

  k_fill<<<1, B256, 0, stream>>>(zerob, 0.0f, 256);
  k_edges<<<(ETOT + 255) / 256, B256, 0, stream>>>(ei, src, dst);

  GEMM(x, ip_w, ip_b, t2, NNODE, HID, 128);
  k_ln_act<<<NNODE / 8, B256, 0, stream>>>(t2, ip_g, ip_bb, h, nullptr, 0, 0, 1);

  for (int i = 0; i < 4; ++i) {
    GEMM(h, blk[i].resw, zerob, res, NNODE, HID, HID);
    GEMM(h, blk[i].wl, blk[i].bl, xl, NNODE, HID, HID);
    GEMM(h, blk[i].wr, blk[i].br, xr, NNODE, HID, HID);
    k_fill<<<(NNODE * GHEADS + 255) / 256, B256, 0, stream>>>((float*)menc, 0.0f, NNODE * GHEADS);
    k_fill<<<(NNODE * GHEADS + 255) / 256, B256, 0, stream>>>(denom, 0.0f, NNODE * GHEADS);
    k_fill<<<(NNODE * HID + 255) / 256, B256, 0, stream>>>(gacc, 0.0f, NNODE * HID);
    k_gat_logits<<<(ETOT * GHEADS + 255) / 256, B256, 0, stream>>>(xl, xr, blk[i].att, src, dst, logit, menc);
    k_gat_expsum<<<(ETOT * GHEADS + 255) / 256, B256, 0, stream>>>(logit, menc, dst, ae, denom);
    k_gat_scatter<<<ETOT, B256, 0, stream>>>(ae, xl, src, dst, gacc);
    k_gat_final<<<(NNODE * HID + 255) / 256, B256, 0, stream>>>(gacc, denom, blk[i].bias, t3);
    k_colmean<<<HID, B256, 0, stream>>>(t3, mean);
    k_se_mlp<<<1, B256, 0, stream>>>(mean, blk[i].se1, blk[i].se2, gate);
    k_se_res<<<(NNODE * HID + 255) / 256, B256, 0, stream>>>(t3, gate, res, t2);
    k_ln_act<<<NNODE / 8, B256, 0, stream>>>(t2, blk[i].g, blk[i].b, h, jk, 1024, i * 256, 2);
  }

  // self-attention
  k_ln_act<<<NNODE / 8, B256, 0, stream>>>(h, ln1g, ln1b, t2, nullptr, 0, 0, 0);
  GEMM(t2, wq, bq, qb, NNODE, HID, HID);
  GEMM(t2, wk, bk, kb2, NNODE, HID, HID);
  GEMM(t2, wv, bvv, vb, NNODE, HID, HID);
  k_flash<<<dim3(NNODE / 64, SAHEADS), dim3(128), 0, stream>>>(qb, kb2, vb, ctx);
  GEMM(ctx, wo, bo, t3, NNODE, HID, HID);
  k_add2<<<(NNODE * HID + 255) / 256, B256, 0, stream>>>(t3, h, t2, HID, 0, NNODE * HID);
  k_ln_act<<<NNODE / 8, B256, 0, stream>>>(t2, ln2g, ln2b, t3, nullptr, 0, 0, 0);
  GEMM(t3, fw1, fb1, t1, NNODE, 512, HID);
  k_act<<<(NNODE * 512 + 255) / 256, B256, 0, stream>>>(t1, t1, NNODE * 512, 1);
  GEMM(t1, fw2, fb2, t3, NNODE, HID, 512);
  k_add2<<<(NNODE * HID + 255) / 256, B256, 0, stream>>>(t3, t2, jk, 1024, 3 * 256, NNODE * HID);

  // MLP head
  GEMM(jk, mw1, mb1, t2, NNODE, HID, 1024);
  k_ln_act<<<NNODE / 8, B256, 0, stream>>>(t2, mg, mb, t3, nullptr, 0, 0, 1);
  GEMM(t3, mw2, mb2, z2, NNODE, 128, HID);
  k_act<<<(NNODE * 128 + 255) / 256, B256, 0, stream>>>(z2, z2, NNODE * 128, 1);
  k_head<<<dim3(12, NNODE / 256), B256, 0, stream>>>(z2, mw3, mb3, (float*)d_out);
}
